// MoE_31507880084033
// MI455X (gfx1250) — matrix operand order repaired and checked by reading, not compiled, not run on hardware
//
#include <hip/hip_runtime.h>
#include <hip/hip_bf16.h>

// Problem constants (match reference)
#define B_    64
#define E_    16
#define C_    1024
#define MID_  4096
#define TOPK_ 8
#define LN_EPS_ 1e-6f

// Padded K-stride (in bf16 elements) for the transposed [N][K] LDS tile.
// 40 elems = 80 B: keeps 16 B alignment for b128 ops, reduces bank conflicts.
#define LDSK_ 40

typedef __attribute__((ext_vector_type(16))) __bf16 v16bf;
typedef __attribute__((ext_vector_type(8)))  float  v8f;

union FragBF { uint4 q[2]; v16bf v; };
union FragA8 { unsigned u[8]; v16bf v; };

__device__ __forceinline__ float qgelu(float v) {
  return v / (1.0f + __expf(-1.702f * v));   // x * sigmoid(1.702 x)
}

__device__ __forceinline__ unsigned pack_bf16(float a, float b) {
  union { __bf16 h[2]; unsigned u; } r;
  r.h[0] = (__bf16)a;   // bits [15:0]
  r.h[1] = (__bf16)b;   // bits [31:16]
  return r.u;           // lowers to v_cvt_pk_bf16_f32
}

// ---------------------------------------------------------------------------
// Kernel 1: router. gate_sum[b][e] = sum_c x[b,e,c]; sel[b][e] = in top-8.
// ---------------------------------------------------------------------------
__global__ void moe_router(const float* __restrict__ x,
                           float* __restrict__ gate_sum,
                           int* __restrict__ selflag) {
  const int b   = blockIdx.x;
  const int tid = threadIdx.x;
  __shared__ float red[256];
  __shared__ float score[E_];

  for (int e = 0; e < E_; ++e) {
    const float* row = x + ((size_t)b * E_ + e) * C_;
    float s = 0.f;
    for (int c = tid; c < C_; c += 256) s += row[c];
    red[tid] = s;
    __syncthreads();
    for (int off = 128; off > 0; off >>= 1) {
      if (tid < off) red[tid] += red[tid + off];
      __syncthreads();
    }
    if (tid == 0) score[e] = red[0];
    __syncthreads();
  }

  if (tid < E_) {
    const int e = tid;
    const float se = score[e];
    int rank = 0;
    for (int j = 0; j < E_; ++j) {
      const float sj = score[j];
      if (sj > se || (sj == se && j < e)) rank++;   // jax top_k tie order
    }
    gate_sum[b * E_ + e] = se;
    selflag[b * E_ + e]  = (rank < TOPK_) ? 1 : 0;
  }
}

// ---------------------------------------------------------------------------
// Kernel 2: h1[b,e,m] = qgelu( x[b,e,:] @ W1[e,:,m] + b1[e,m] ), bf16 out.
// Grid: (MID/64, E). Block 256 = 8 waves. M=64 (batch), N tile = 64, K-step 32.
// W1 tile staged once per block into transposed bf16 LDS (no redundant HBM).
// ---------------------------------------------------------------------------
__global__ void moe_gemm1(const float* __restrict__ x,
                          const float* __restrict__ W1,
                          const float* __restrict__ b1,
                          __bf16* __restrict__ h1) {
  __shared__ __align__(16) unsigned short smem[64 * LDSK_];   // [N=64][K pad 40]

  const int e    = blockIdx.y;
  const int N0   = blockIdx.x * 64;
  const int tid  = threadIdx.x;
  const int w    = tid >> 5;
  const int lane = tid & 31;
  const int half = lane >> 4;      // K-half select (ISA 16-bit operand layout)
  const int lidx = lane & 15;      // M row (A) / N col (B) within 16-tile
  const int m0   = (w & 3) * 16;   // batch-row base
  const int n0   = (w >> 2) * 32;  // LDS-local N base (two 16-wide tiles)

  // staging coords: thread owns column sn, 8 consecutive K rows at sk
  const int sn = tid & 63;
  const int sk = (tid >> 6) * 8;

  v8f acc0 = {}, acc1 = {};
  const float* arow = x + ((size_t)(m0 + lidx) * E_ + e) * C_ + half * 8;
  const float* wcol = W1 + (size_t)e * C_ * MID_ + (size_t)sk * MID_ + N0 + sn;

  for (int k0 = 0; k0 < C_; k0 += 32) {
    // ---- coalesced NT fetch of this thread's W slice (each elem read once) --
    float f[8];
#pragma unroll
    for (int j = 0; j < 8; ++j)
      f[j] = __builtin_nontemporal_load(wcol + (size_t)(k0 + j) * MID_);

    // ---- A fragment: aligned float4 loads from L2-resident x, pack to bf16 --
    float4 a0 = *(const float4*)(arow + k0);
    float4 a1 = *(const float4*)(arow + k0 + 4);
    float4 a2 = *(const float4*)(arow + k0 + 16);
    float4 a3 = *(const float4*)(arow + k0 + 20);
    FragA8 au;
    au.u[0] = pack_bf16(a0.x, a0.y);
    au.u[1] = pack_bf16(a0.z, a0.w);
    au.u[2] = pack_bf16(a1.x, a1.y);
    au.u[3] = pack_bf16(a1.z, a1.w);
    au.u[4] = pack_bf16(a2.x, a2.y);
    au.u[5] = pack_bf16(a2.z, a2.w);
    au.u[6] = pack_bf16(a3.x, a3.y);
    au.u[7] = pack_bf16(a3.z, a3.w);

    __syncthreads();   // previous iteration's LDS reads complete
    uint4 pk;
    pk.x = pack_bf16(f[0], f[1]);
    pk.y = pack_bf16(f[2], f[3]);
    pk.z = pack_bf16(f[4], f[5]);
    pk.w = pack_bf16(f[6], f[7]);
    *(uint4*)&smem[sn * LDSK_ + sk] = pk;   // transposed [N][K], one b128 store
    __syncthreads();   // tile visible

    // ---- B fragments: 2x16B contiguous LDS reads per fragment --------------
    FragBF fb0, fb1;
    const uint4* c0 = (const uint4*)&smem[(n0 + lidx) * LDSK_ + half * 8];
    const uint4* c1 = (const uint4*)&smem[(n0 + 16 + lidx) * LDSK_ + half * 8];
    fb0.q[0] = c0[0]; fb0.q[1] = c0[2];
    fb1.q[0] = c1[0]; fb1.q[1] = c1[2];

    acc0 = __builtin_amdgcn_wmma_f32_16x16x32_bf16(false, au.v, false, fb0.v,
                                                   (short)0, acc0, false, false);
    acc1 = __builtin_amdgcn_wmma_f32_16x16x32_bf16(false, au.v, false, fb1.v,
                                                   (short)0, acc1, false, false);
  }

  // C/D layout: VGPR r -> M = r (lanes 0-15) / r+8 (lanes 16-31), N = lane%16.
#pragma unroll
  for (int r = 0; r < 8; ++r) {
    const int m  = m0 + r + half * 8;
    const int na = N0 + n0 + lidx;
    const int nb = N0 + n0 + 16 + lidx;
    float v0 = qgelu(acc0[r] + b1[e * MID_ + na]);
    float v1 = qgelu(acc1[r] + b1[e * MID_ + nb]);
    h1[((size_t)m * E_ + e) * MID_ + na] = (__bf16)v0;
    h1[((size_t)m * E_ + e) * MID_ + nb] = (__bf16)v1;
  }
}

// ---------------------------------------------------------------------------
// Kernel 3: h2[b,e,c] = qgelu( h1[b,e,:] @ W2[e,:,c] + b2[e,c] ), f32 out.
// Same tiling; A operand is already bf16 (pure uint4 loads, no conversion).
// ---------------------------------------------------------------------------
__global__ void moe_gemm2(const __bf16* __restrict__ h1,
                          const float* __restrict__ W2,
                          const float* __restrict__ b2,
                          float* __restrict__ h2) {
  __shared__ __align__(16) unsigned short smem[64 * LDSK_];

  const int e    = blockIdx.y;
  const int N0   = blockIdx.x * 64;
  const int tid  = threadIdx.x;
  const int w    = tid >> 5;
  const int lane = tid & 31;
  const int half = lane >> 4;
  const int lidx = lane & 15;
  const int m0   = (w & 3) * 16;
  const int n0   = (w >> 2) * 32;

  const int sn = tid & 63;
  const int sk = (tid >> 6) * 8;

  v8f acc0 = {}, acc1 = {};
  const __bf16* arow = h1 + ((size_t)(m0 + lidx) * E_ + e) * MID_ + half * 8;
  const float* wcol = W2 + (size_t)e * MID_ * C_ + (size_t)sk * C_ + N0 + sn;

  for (int k0 = 0; k0 < MID_; k0 += 32) {
    float f[8];
#pragma unroll
    for (int j = 0; j < 8; ++j)
      f[j] = __builtin_nontemporal_load(wcol + (size_t)(k0 + j) * C_);

    FragBF au;   // A already bf16: two aligned 16B loads
    au.q[0] = *(const uint4*)(arow + k0);
    au.q[1] = *(const uint4*)(arow + k0 + 16);

    __syncthreads();
    uint4 pk;
    pk.x = pack_bf16(f[0], f[1]);
    pk.y = pack_bf16(f[2], f[3]);
    pk.z = pack_bf16(f[4], f[5]);
    pk.w = pack_bf16(f[6], f[7]);
    *(uint4*)&smem[sn * LDSK_ + sk] = pk;
    __syncthreads();

    FragBF fb0, fb1;
    const uint4* c0 = (const uint4*)&smem[(n0 + lidx) * LDSK_ + half * 8];
    const uint4* c1 = (const uint4*)&smem[(n0 + 16 + lidx) * LDSK_ + half * 8];
    fb0.q[0] = c0[0]; fb0.q[1] = c0[2];
    fb1.q[0] = c1[0]; fb1.q[1] = c1[2];

    acc0 = __builtin_amdgcn_wmma_f32_16x16x32_bf16(false, au.v, false, fb0.v,
                                                   (short)0, acc0, false, false);
    acc1 = __builtin_amdgcn_wmma_f32_16x16x32_bf16(false, au.v, false, fb1.v,
                                                   (short)0, acc1, false, false);
  }

#pragma unroll
  for (int r = 0; r < 8; ++r) {
    const int m  = m0 + r + half * 8;
    const int na = N0 + n0 + lidx;
    const int nb = N0 + n0 + 16 + lidx;
    h2[((size_t)m * E_ + e) * C_ + na] = qgelu(acc0[r] + b2[e * C_ + na]);
    h2[((size_t)m * E_ + e) * C_ + nb] = qgelu(acc1[r] + b2[e * C_ + nb]);
  }
}

// ---------------------------------------------------------------------------
// Kernel 4: LayerNorm of selected experts + gate-weighted mean over top-k.
// ---------------------------------------------------------------------------
__global__ void moe_combine(const float* __restrict__ h2,
                            const float* __restrict__ gate_sum,
                            const int* __restrict__ selflag,
                            const float* __restrict__ ln_g,
                            const float* __restrict__ ln_b,
                            float* __restrict__ out) {
  const int b   = blockIdx.x;
  const int tid = threadIdx.x;
  __shared__ float red[256];

  float acc[C_ / 256];
#pragma unroll
  for (int i = 0; i < C_ / 256; ++i) acc[i] = 0.f;

  for (int e = 0; e < E_; ++e) {
    if (!selflag[b * E_ + e]) continue;  // uniform across the block
    const float* row = h2 + ((size_t)b * E_ + e) * C_;

    float s1 = 0.f, s2 = 0.f;
#pragma unroll
    for (int i = 0; i < C_ / 256; ++i) {
      const float v = row[tid + 256 * i];
      s1 += v;
      s2 += v * v;
    }
    red[tid] = s1;
    __syncthreads();
    for (int off = 128; off > 0; off >>= 1) {
      if (tid < off) red[tid] += red[tid + off];
      __syncthreads();
    }
    const float S1 = red[0];
    __syncthreads();
    red[tid] = s2;
    __syncthreads();
    for (int off = 128; off > 0; off >>= 1) {
      if (tid < off) red[tid] += red[tid + off];
      __syncthreads();
    }
    const float S2 = red[0];
    __syncthreads();

    const float mu  = S1 * (1.0f / C_);
    const float var = S2 * (1.0f / C_) - mu * mu;
    const float rs  = rsqrtf(var + LN_EPS_);
    const float g   = gate_sum[b * E_ + e];

#pragma unroll
    for (int i = 0; i < C_ / 256; ++i) {
      const int c = tid + 256 * i;
      const float v = row[c];
      acc[i] += g * ((v - mu) * rs * ln_g[e * C_ + c] + ln_b[e * C_ + c]);
    }
  }

#pragma unroll
  for (int i = 0; i < C_ / 256; ++i)
    out[(size_t)b * C_ + tid + 256 * i] = acc[i] * (1.0f / TOPK_);
}

// ---------------------------------------------------------------------------
// Host launcher
// ---------------------------------------------------------------------------
extern "C" void kernel_launch(void* const* d_in, const int* in_sizes, int n_in,
                              void* d_out, int out_size, void* d_ws, size_t ws_size,
                              hipStream_t stream) {
  (void)in_sizes; (void)n_in; (void)out_size; (void)ws_size;
  const float* x   = (const float*)d_in[0];
  // d_in[1] = q : unused by the reference output
  const float* W1  = (const float*)d_in[2];
  const float* b1  = (const float*)d_in[3];
  const float* W2  = (const float*)d_in[4];
  const float* b2  = (const float*)d_in[5];
  const float* lng = (const float*)d_in[6];
  const float* lnb = (const float*)d_in[7];
  float* out = (float*)d_out;

  // Workspace layout
  char* ws = (char*)d_ws;
  float*  gate_sum = (float*)ws;                                        // 4 KB
  int*    selflag  = (int*)(ws + 4096);                                 // 4 KB
  __bf16* h1       = (__bf16*)(ws + 8192);                              // 8 MB
  float*  h2       = (float*)(ws + 8192 + (size_t)B_ * E_ * MID_ * 2);  // 4 MB

  moe_router <<<B_, 256, 0, stream>>>(x, gate_sum, selflag);
  moe_gemm1  <<<dim3(MID_ / 64, E_), 256, 0, stream>>>(x, W1, b1, h1);
  moe_gemm2  <<<dim3(C_ / 64, E_), 256, 0, stream>>>(h1, W2, b2, h2);
  moe_combine<<<B_, 256, 0, stream>>>(h2, gate_sum, selflag, lng, lnb, out);
}
